// Head_46763603918940
// MI455X (gfx1250) — compile-verified, matrix operand order checked by reading, not run on hardware
//
#include <hip/hip_runtime.h>

// Problem constants (match reference)
#define BB  16
#define LL  2048
#define EE  1024
#define DKK 128

typedef __bf16 bf16_t;
typedef __attribute__((ext_vector_type(8)))  bf16_t v8bf;
typedef __attribute__((ext_vector_type(16))) bf16_t v16bf;
typedef __attribute__((ext_vector_type(8)))  float  v8f;

// Load a 16x32 (rows x K) bf16 WMMA A/B fragment from a row-major tile.
// `p` = base + row*ld + kbase. Per ISA: lane group g=lane>>4 owns k in
// [g*8, g*8+8) and [16+g*8, 16+g*8+8) -> two contiguous 16B chunks.
__device__ __forceinline__ v16bf frag_ld(const bf16_t* __restrict__ p, int g) {
  const bf16_t* q = p + g * 8;
  v8bf lo = *(const v8bf*)(q);
  v8bf hi = *(const v8bf*)(q + 16);
  return __builtin_shufflevector(lo, hi, 0,1,2,3,4,5,6,7,8,9,10,11,12,13,14,15);
}

__device__ __forceinline__ v16bf frag_ld_lds(const bf16_t* p, int g) {
  const bf16_t* q = p + g * 8;
  v8bf lo = *(const v8bf*)(q);
  v8bf hi = *(const v8bf*)(q + 16);
  return __builtin_shufflevector(lo, hi, 0,1,2,3,4,5,6,7,8,9,10,11,12,13,14,15);
}

// ---------------------------------------------------------------------------
// Kernel 1: fused projection GEMM.  grid.y selects matrix (0:Q 1:K 2:V).
// Block = 128 threads = 4 waves; each wave computes a 16-token x 128 tile.
// W slice (32x128) staged transposed in LDS as bf16 each K-step.
// Q,K stored row-major bf16 [B,L,DK]; V stored transposed [B,DK,L].
// Length mask applied on store (zero padded token rows), matching reference.
// ---------------------------------------------------------------------------
__global__ __launch_bounds__(128) void proj_kernel(
    const float* __restrict__ x,  const float* __restrict__ ctx,
    const float* __restrict__ Wq, const float* __restrict__ Wk,
    const float* __restrict__ Wv, const int* __restrict__ lens,
    bf16_t* __restrict__ Qb, bf16_t* __restrict__ Kb, bf16_t* __restrict__ Vt)
{
  __shared__ __align__(16) bf16_t Wt[128][32];   // [n][k] transposed slice

  const int tid  = threadIdx.x;
  const int w    = tid >> 5;
  const int lane = tid & 31;
  const int ln   = lane & 15;
  const int g    = lane >> 4;
  const int mat  = blockIdx.y;                    // 0:Q 1:K 2:V
  const float* __restrict__ src = (mat == 2) ? x : ctx;
  const float* __restrict__ W   = (mat == 0) ? Wq : (mat == 1 ? Wk : Wv);

  const int t0   = blockIdx.x * 64;               // first token of block
  const int rowA = t0 + w * 16 + ln;              // this lane's A-matrix row

  v8f c[8];
  #pragma unroll
  for (int t = 0; t < 8; ++t) c[t] = (v8f){};

  for (int ks = 0; ks < EE / 32; ++ks) {
    const int kb = ks * 32;
    // Stage W[kb..kb+31][0..127] transposed -> Wt[n][k] (f32 -> bf16).
    #pragma unroll
    for (int idx = tid; idx < 32 * 128; idx += 128) {
      const int kk = idx >> 7;
      const int nn = idx & 127;
      Wt[nn][kk] = (bf16_t)W[(size_t)(kb + kk) * DKK + nn];
    }
    __syncthreads();

    // A fragment straight from global f32 (two contiguous 32B chunks) + cvt.
    const float* pa = src + (size_t)rowA * EE + kb + g * 8;
    v16bf a;
    #pragma unroll
    for (int i = 0; i < 8; ++i) a[i] = (bf16_t)pa[i];
    #pragma unroll
    for (int i = 0; i < 8; ++i) a[8 + i] = (bf16_t)pa[16 + i];

    #pragma unroll
    for (int t = 0; t < 8; ++t) {
      v16bf bfr = frag_ld_lds(&Wt[t * 16 + ln][0], g);
      c[t] = __builtin_amdgcn_wmma_f32_16x16x32_bf16(
                 false, a, false, bfr, (short)0, c[t], false, false);
    }
    __syncthreads();
  }

  // Epilogue: length-mask rows, store bf16.
  #pragma unroll
  for (int r = 0; r < 8; ++r) {
    const int tok = t0 + w * 16 + r + 8 * g;      // C layout: M = r + 8*g
    const int bb  = tok >> 11;                    // / LL
    const int lt  = tok & (LL - 1);
    const bool valid = lt < lens[bb];
    #pragma unroll
    for (int t = 0; t < 8; ++t) {
      const float v = valid ? c[t][r] : 0.0f;
      const int n = t * 16 + ln;
      if (mat == 0)      Qb[(size_t)tok * DKK + n] = (bf16_t)v;
      else if (mat == 1) Kb[(size_t)tok * DKK + n] = (bf16_t)v;
      else               Vt[((size_t)bb * DKK + n) * LL + lt] = (bf16_t)v;
    }
  }
}

// ---------------------------------------------------------------------------
// Kernel 2: causal flash attention, one wave per (batch, 16-query tile).
// Online softmax over 32-key blocks; S and P*V via bf16 WMMA, f32 accum.
// ---------------------------------------------------------------------------
__global__ __launch_bounds__(128) void attn_kernel(
    const bf16_t* __restrict__ Qb, const bf16_t* __restrict__ Kb,
    const bf16_t* __restrict__ Vt, const int* __restrict__ lens,
    float* __restrict__ out)
{
  __shared__ __align__(16) bf16_t Pst[4][16][32];  // per-wave P staging

  const int tid  = threadIdx.x;
  const int w    = tid >> 5;
  const int lane = tid & 31;
  const int ln   = lane & 15;
  const int g    = lane >> 4;

  const int gw    = blockIdx.x * 4 + w;            // global wave id
  const int b     = gw >> 7;                       // / (LL/16)
  const int qt    = gw & 127;
  const int qBase = qt * 16;
  const int len   = lens[b];

  const bf16_t* __restrict__ Qp = Qb + (size_t)b * LL * DKK;
  const bf16_t* __restrict__ Kp = Kb + (size_t)b * LL * DKK;
  const bf16_t* __restrict__ Vp = Vt + (size_t)b * DKK * LL;

  // Q tile fragments (16 queries x 128), resident for the whole loop.
  v16bf qa[4];
  #pragma unroll
  for (int kk = 0; kk < 4; ++kk)
    qa[kk] = frag_ld(Qp + (size_t)(qBase + ln) * DKK + kk * 32, g);

  v8f O[8];
  float M[8], Lr[8];
  #pragma unroll
  for (int t = 0; t < 8; ++t) O[t] = (v8f){};
  #pragma unroll
  for (int r = 0; r < 8; ++r) { M[r] = -__builtin_inff(); Lr[r] = 0.0f; }

  const int keyEnd = min(qBase + 16, len);         // causal + length skip
  const int nblk   = (keyEnd + 31) >> 5;
  const float scale = 0.08838834764831845f;        // 1/sqrt(128)

  for (int kb = 0; kb < nblk; ++kb) {
    const int k0 = kb * 32;
    if (kb + 1 < nblk) {                           // prefetch next K block
      __builtin_prefetch(Kp + (size_t)(k0 + 32 + ln) * DKK, 0, 0);
      __builtin_prefetch(Kp + (size_t)(k0 + 48 + ln) * DKK, 0, 0);
    }

    // S = Q * K^T : two 16x16 column tiles (keys k0..15, k0+16..31).
    v8f s0 = (v8f){}, s1 = (v8f){};
    #pragma unroll
    for (int kk = 0; kk < 4; ++kk) {
      v16bf bk = frag_ld(Kp + (size_t)(k0 + ln) * DKK + kk * 32, g);
      s0 = __builtin_amdgcn_wmma_f32_16x16x32_bf16(
               false, qa[kk], false, bk, (short)0, s0, false, false);
    }
    #pragma unroll
    for (int kk = 0; kk < 4; ++kk) {
      v16bf bk = frag_ld(Kp + (size_t)(k0 + 16 + ln) * DKK + kk * 32, g);
      s1 = __builtin_amdgcn_wmma_f32_16x16x32_bf16(
               false, qa[kk], false, bk, (short)0, s1, false, false);
    }

    // Online softmax update (rows m = r + 8*g, cols = ln / 16+ln).
    float alpha_[8];
    #pragma unroll
    for (int r = 0; r < 8; ++r) {
      const int qi = qBase + r + 8 * g;
      const int j0 = k0 + ln;
      const int j1 = k0 + 16 + ln;
      const float e0 = (j0 <= qi && j0 < len) ? s0[r] * scale : -__builtin_inff();
      const float e1 = (j1 <= qi && j1 < len) ? s1[r] * scale : -__builtin_inff();
      float tmx = fmaxf(e0, e1);
      #pragma unroll
      for (int ms = 1; ms <= 8; ms <<= 1)
        tmx = fmaxf(tmx, __shfl_xor(tmx, ms, 32));
      const float Mn = fmaxf(M[r], tmx);           // finite: col j=k0 always valid
      const float al = __expf(M[r] - Mn);
      const float p0 = __expf(e0 - Mn);
      const float p1 = __expf(e1 - Mn);
      float rs = p0 + p1;
      #pragma unroll
      for (int ms = 1; ms <= 8; ms <<= 1)
        rs += __shfl_xor(rs, ms, 32);
      Lr[r] = Lr[r] * al + rs;
      M[r]  = Mn;
      alpha_[r] = al;
      // Stash P tile (bf16) in LDS in C layout for re-read as A fragment.
      Pst[w][r + 8 * g][ln]      = (bf16_t)p0;
      Pst[w][r + 8 * g][16 + ln] = (bf16_t)p1;
    }
    #pragma unroll
    for (int t = 0; t < 8; ++t) {
      #pragma unroll
      for (int r = 0; r < 8; ++r) O[t][r] *= alpha_[r];
    }

    // Cross-lane LDS RAW inside the wave: wait for the ds stores.
    asm volatile("s_wait_dscnt 0" ::: "memory");
    v16bf pf = frag_ld_lds(&Pst[w][ln][0], g);

    // O += P * V ; V is pre-transposed so B fragments are contiguous loads.
    #pragma unroll
    for (int t = 0; t < 8; ++t) {
      v16bf vf = frag_ld(Vp + (size_t)(t * 16 + ln) * LL + k0, g);
      O[t] = __builtin_amdgcn_wmma_f32_16x16x32_bf16(
                 false, pf, false, vf, (short)0, O[t], false, false);
    }
  }

  // Epilogue: normalize and store f32 output [B, L, DK].
  #pragma unroll
  for (int r = 0; r < 8; ++r) {
    const float inv = 1.0f / Lr[r];
    const int q = qBase + r + 8 * g;
    #pragma unroll
    for (int t = 0; t < 8; ++t)
      out[((size_t)(b * LL + q)) * DKK + t * 16 + ln] = O[t][r] * inv;
  }
}

// ---------------------------------------------------------------------------
extern "C" void kernel_launch(void* const* d_in, const int* in_sizes, int n_in,
                              void* d_out, int out_size, void* d_ws, size_t ws_size,
                              hipStream_t stream) {
  const float* x    = (const float*)d_in[0];
  const float* ctx  = (const float*)d_in[1];
  const int*   lens = (const int*)d_in[2];
  const float* Wq   = (const float*)d_in[3];
  const float* Wk   = (const float*)d_in[4];
  const float* Wv   = (const float*)d_in[5];
  float* out = (float*)d_out;

  // Workspace: Q,K row-major bf16 + V transposed bf16 (3 * 8 MB = 24 MB).
  bf16_t* Qb = (bf16_t*)d_ws;
  bf16_t* Kb = Qb + (size_t)BB * LL * DKK;
  bf16_t* Vt = Kb + (size_t)BB * LL * DKK;

  dim3 g1(BB * LL / 64, 3), b1(128);
  proj_kernel<<<g1, b1, 0, stream>>>(x, ctx, Wq, Wk, Wv, lens, Qb, Kb, Vt);

  dim3 g2((BB * LL / 16) / 4), b2(128);
  attn_kernel<<<g2, b2, 0, stream>>>(Qb, Kb, Vt, lens, out);
}
